// GungnirHalfKA_53317724012566
// MI455X (gfx1250) — compile-verified, matrix operand order checked by reading, not run on hardware
//
#include <hip/hip_runtime.h>

#define FT_IN   22528
#define FT_OUT  1024
#define HHALF   512
#define NB      8192
#define TILE    16
#define FEATS   32
#define STACKS  8
#define FTH_STRIDE 1032   // f16 elements; 2064B row stride -> conflict-free b128 LDS reads

typedef __attribute__((ext_vector_type(16))) _Float16 v16h;
typedef __attribute__((ext_vector_type(8)))  _Float16 v8h;
typedef __attribute__((ext_vector_type(8)))  float    v8f;

union HFrag { v16h v; v8h h[2]; };

__device__ __forceinline__ float crelu127(float x) {
  return fminf(fmaxf(x, 0.0f), 127.0f);
}

// ---------------------------------------------------------------------------
// Kernel 0: swizzle fc0_w (f32 [stack][out=16][h=1024]) into f16 WMMA B-frags.
// Fragment layout: frag[s][chunk][lane][i], i in 0..15 per lane (v16h).
// B-matrix (K x N) 16-bit layout: lanes 0-15 hold N=lane, K in {0..7,16..23};
// lanes 16-31 hold N=lane-16, K in {8..15,24..31}; elements 0..7 -> first K
// octet, 8..15 -> second (+16) octet.  B[k][n] = fc0_w[s][n][k].
// ---------------------------------------------------------------------------
__global__ __launch_bounds__(256)
void nnue_swizzle_fc0(const float* __restrict__ fc0_w, _Float16* __restrict__ frag) {
  const int gid   = blockIdx.x * 256 + threadIdx.x;   // 131072 total
  const int i     = gid & 15;
  const int lane  = (gid >> 4) & 31;
  const int chunk = (gid >> 9) & 31;
  const int s     = gid >> 14;
  const int n     = lane & 15;
  const int khalf = lane >> 4;
  const int k_in  = (i < 8) ? (khalf * 8 + i) : (16 + khalf * 8 + (i - 8));
  const int k     = chunk * 32 + k_in;
  frag[gid] = (_Float16)fc0_w[(size_t)(s * 16 + n) * FT_OUT + k];
}

// ---------------------------------------------------------------------------
// Kernel 1: fused gather -> pairwise -> WMMA fc0 -> fc1/fc2 tail.
// One block = 16 samples.  256 threads = 8 waves (wave32).
// ---------------------------------------------------------------------------
__global__ __launch_bounds__(256)
void nnue_fused(const int*   __restrict__ w_feats,
                const int*   __restrict__ b_feats,
                const int*   __restrict__ stm,
                const int*   __restrict__ bucket,
                const float* __restrict__ ft_w,
                const float* __restrict__ ft_bias,
                const float* __restrict__ psqt_w,
                const _Float16* __restrict__ fc0_frag,
                const float* __restrict__ fc0_b,
                const float* __restrict__ fc1_w,
                const float* __restrict__ fc1_b,
                const float* __restrict__ fc2_w,
                const float* __restrict__ fc2_b,
                float* __restrict__ out) {
  __shared__ _Float16 ft_h[TILE * FTH_STRIDE];          // ~33 KB  f16 ft tile
  __shared__ float    o0a[STACKS][TILE * 16];           // 8 KB    fc0 out, all stacks
  __shared__ float    psqts[TILE];

  const int tile = blockIdx.x;
  const int tid  = threadIdx.x;

  // ---------------- Phase A: embedding-bag gather + pairwise -> LDS (f16) ---
  {
    const int smp0 = tid >> 7;          // 0 or 1
    const int c    = (tid & 127) * 4;   // column-pair group: cols c..c+3 & c+512..c+515
    const float4 bia_lo = *(const float4*)(ft_bias + c);
    const float4 bia_hi = *(const float4*)(ft_bias + c + HHALF);
#pragma unroll 1
    for (int rep = 0; rep < 8; ++rep) {
      const int smp = smp0 + rep * 2;
      const int gs  = tile * TILE + smp;
      float aw[8], ab[8];
#pragma unroll
      for (int j = 0; j < 8; ++j) { aw[j] = 0.0f; ab[j] = 0.0f; }

      const int* __restrict__ wf = w_feats + (size_t)gs * FEATS;
      const int* __restrict__ bf = b_feats + (size_t)gs * FEATS;
#pragma unroll 1
      for (int f = 0; f < FEATS; ++f) {
        const size_t row = (size_t)wf[f] * FT_OUT;
        if (f + 1 < FEATS)
          __builtin_prefetch(ft_w + (size_t)wf[f + 1] * FT_OUT + c, 0, 0);
        const float4 lo = *(const float4*)(ft_w + row + c);
        const float4 hi = *(const float4*)(ft_w + row + c + HHALF);
        aw[0] += lo.x; aw[1] += lo.y; aw[2] += lo.z; aw[3] += lo.w;
        aw[4] += hi.x; aw[5] += hi.y; aw[6] += hi.z; aw[7] += hi.w;
      }
#pragma unroll 1
      for (int f = 0; f < FEATS; ++f) {
        const size_t row = (size_t)bf[f] * FT_OUT;
        if (f + 1 < FEATS)
          __builtin_prefetch(ft_w + (size_t)bf[f + 1] * FT_OUT + c, 0, 0);
        const float4 lo = *(const float4*)(ft_w + row + c);
        const float4 hi = *(const float4*)(ft_w + row + c + HHALF);
        ab[0] += lo.x; ab[1] += lo.y; ab[2] += lo.z; ab[3] += lo.w;
        ab[4] += hi.x; ab[5] += hi.y; ab[6] += hi.z; ab[7] += hi.w;
      }
      // + ft_bias (applied to both accumulators, as in reference)
      aw[0] += bia_lo.x; aw[1] += bia_lo.y; aw[2] += bia_lo.z; aw[3] += bia_lo.w;
      aw[4] += bia_hi.x; aw[5] += bia_hi.y; aw[6] += bia_hi.z; aw[7] += bia_hi.w;
      ab[0] += bia_lo.x; ab[1] += bia_lo.y; ab[2] += bia_lo.z; ab[3] += bia_lo.w;
      ab[4] += bia_hi.x; ab[5] += bia_hi.y; ab[6] += bia_hi.z; ab[7] += bia_hi.w;

      const int st = stm[gs];
      _Float16* __restrict__ rowp = ft_h + smp * FTH_STRIDE;
#pragma unroll
      for (int j = 0; j < 4; ++j) {
        const float s_lo = st ? ab[j]     : aw[j];
        const float s_hi = st ? ab[4 + j] : aw[4 + j];
        const float o_lo = st ? aw[j]     : ab[j];
        const float o_hi = st ? aw[4 + j] : ab[4 + j];
        const float ps = crelu127(s_lo) * crelu127(s_hi) * (1.0f / 128.0f);
        const float po = crelu127(o_lo) * crelu127(o_hi) * (1.0f / 128.0f);
        rowp[c + j]          = (_Float16)ps;   // ft[0..511]  from acc_stm
        rowp[HHALF + c + j]  = (_Float16)po;   // ft[512..1023] from acc_opp
      }
    }
  }

  // ---------------- psqt term per sample (threads 0..127) -------------------
  if (tid < 128) {
    const int smp = tid >> 3;
    const int pb  = tid & 7;
    const int gs  = tile * TILE + smp;
    float pwv = 0.0f, pbv = 0.0f;
    const int* __restrict__ wf = w_feats + (size_t)gs * FEATS;
    const int* __restrict__ bf = b_feats + (size_t)gs * FEATS;
    for (int f = 0; f < FEATS; ++f) {
      pwv += psqt_w[(size_t)wf[f] * STACKS + pb];
      pbv += psqt_w[(size_t)bf[f] * STACKS + pb];
    }
    const int st = stm[gs];
    if (pb == bucket[gs]) {
      const float p_stm = st ? pbv : pwv;
      const float p_opp = st ? pwv : pbv;
      psqts[smp] = (p_stm - p_opp) * 0.5f;
    }
  }
  __syncthreads();

  // ---------------- Phase B: fc0 via WMMA, one wave per bucket stack --------
  {
    const int s     = tid >> 5;        // stack / wave id
    const int lane  = tid & 31;
    const int lidx  = lane & 15;       // A: M row;  C/D: N column
    const int khalf = lane >> 4;

    v8f acc = {};
    const v8h* __restrict__ bfr =
        (const v8h*)(fc0_frag) + ((size_t)(s * 32) * 32 + lane) * 2;
    const _Float16* __restrict__ abase = &ft_h[lidx * FTH_STRIDE + khalf * 8];
#pragma unroll 4
    for (int chunk = 0; chunk < 32; ++chunk) {
      const _Float16* arow = abase + chunk * 32;
      HFrag a, b;
      a.h[0] = *(const v8h*)(arow);        // K octet (khalf ? 8..15 : 0..7)
      a.h[1] = *(const v8h*)(arow + 16);   // K octet +16
      b.h[0] = bfr[(size_t)chunk * 64 + 0];
      b.h[1] = bfr[(size_t)chunk * 64 + 1];
      acc = __builtin_amdgcn_wmma_f32_16x16x32_f16(
          false, a.v, false, b.v, (short)0, acc, false, false);
    }

    // unconditional write of the whole stack tile; tail selects by bucket
    const float biasn = fc0_b[s * 16 + lidx];
#pragma unroll
    for (int r = 0; r < 8; ++r) {
      const int ms = r + khalf * 8;               // sample row in tile
      o0a[s][ms * 16 + lidx] = acc[r] + biasn;
    }
  }
  __syncthreads();

  // ---------------- Phase C: fc1 / fc2 / skip / output ----------------------
  if (tid < TILE) {
    const int smp = tid;
    const int gs  = tile * TILE + smp;
    const int bk  = bucket[gs];

    float o0v[16];
#pragma unroll
    for (int n = 0; n < 16; ++n) o0v[n] = o0a[bk][smp * 16 + n];

    float slab[32];
#pragma unroll
    for (int i = 0; i < 15; ++i) {
      const float q = o0v[i];
      slab[i]      = fminf(fmaxf(q * q * (1.0f / 524288.0f), 0.0f), 127.0f);
      slab[15 + i] = crelu127(q * (1.0f / 64.0f));
    }
    slab[30] = 0.0f; slab[31] = 0.0f;

    const float* __restrict__ w1 = fc1_w + (size_t)bk * 32 * 32;
    float sc = fc2_b[bk];
#pragma unroll 1
    for (int o = 0; o < 32; ++o) {
      float s1 = fc1_b[bk * 32 + o];
#pragma unroll
      for (int i = 0; i < 32; ++i) s1 += slab[i] * w1[o * 32 + i];
      sc += crelu127(s1 * (1.0f / 64.0f)) * fc2_w[bk * 32 + o];
    }
    const float skip = o0v[15] * (9600.0f / 8128.0f);
    out[gs] = (psqts[smp] + sc + skip) * (1.0f / 16.0f);
  }
}

extern "C" void kernel_launch(void* const* d_in, const int* in_sizes, int n_in,
                              void* d_out, int out_size, void* d_ws, size_t ws_size,
                              hipStream_t stream) {
  const int*   w_feats = (const int*)d_in[0];
  // d_in[1] = w_offsets (uniform stride 32, unused)
  const int*   b_feats = (const int*)d_in[2];
  // d_in[3] = b_offsets (unused)
  const int*   stm     = (const int*)d_in[4];
  const int*   bucket  = (const int*)d_in[5];
  const float* ft_w    = (const float*)d_in[6];
  const float* ft_bias = (const float*)d_in[7];
  const float* psqt_w  = (const float*)d_in[8];
  const float* fc0_w   = (const float*)d_in[9];
  const float* fc0_b   = (const float*)d_in[10];
  const float* fc1_w   = (const float*)d_in[11];
  const float* fc1_b   = (const float*)d_in[12];
  const float* fc2_w   = (const float*)d_in[13];
  const float* fc2_b   = (const float*)d_in[14];

  _Float16* frag = (_Float16*)d_ws;   // 8*32*32*16 f16 = 256 KB

  nnue_swizzle_fc0<<<512, 256, 0, stream>>>(fc0_w, frag);
  nnue_fused<<<NB / TILE, 256, 0, stream>>>(
      w_feats, b_feats, stm, bucket, ft_w, ft_bias, psqt_w,
      frag, fc0_b, fc1_w, fc1_b, fc2_w, fc2_b, (float*)d_out);
}
